// BiGRUForSegmentation_86268713107547
// MI455X (gfx1250) — matrix operand order repaired and checked by reading, not compiled, not run on hardware
//
#include <hip/hip_runtime.h>
#include <hip/hip_bf16.h>

// BiGRU (2-layer, bidirectional) for MI455X / gfx1250, wave32 + WMMA + TDM.
//
//  1) cvt kernels: W_hh (4x) and layer-1 W_ih (2x) f32 -> f16 in workspace.
//  2) gru_layer0_kernel: persistent, grid = 2 dirs x 4 batch-chunks (16 rows).
//     TDM double-buffers x_t [16x7] f32 tiles into LDS; VALU input gates,
//     WMMA recurrence (h.W_hh^T, B-fragments register-resident), fused gates.
//  3) gru_layer1_kernel: TDM double-buffers x_t [16x256] f16 tiles (with TDM
//     LDS padding for bank-conflict-free fragment reads); input gates AND
//     recurrence via WMMA.
//  4) head_kernel: wave-per-(b,s) dot(gru_out, wout)+bout with shfl reduce.

typedef _Float16 f16;
typedef __attribute__((ext_vector_type(16))) _Float16 v16h;
typedef __attribute__((ext_vector_type(8)))  _Float16 v8h;
typedef __attribute__((ext_vector_type(8)))  float    v8f;
typedef __attribute__((ext_vector_type(4)))  unsigned int v4u;
typedef __attribute__((ext_vector_type(8)))  int v8i;
typedef __attribute__((ext_vector_type(4)))  int v4i;

#define BB   64
#define SS   4096
#define DIN  7
#define HH   128
#define G3H  384
#define C2H  256
#define HS16 136   // padded f16 hidden-state LDS stride (bank shift 4/row)
#define HS32 132   // padded f32 hidden-state LDS stride (bank shift 4/row)
#define XS1  264   // layer-1 staged tile row stride in f16 (256 + TDM pad 16B)

__device__ __forceinline__ v8f wmma_f32_16x16x32_f16(v16h a, v16h b, v8f c) {
  return __builtin_amdgcn_wmma_f32_16x16x32_f16(false, a, false, b, (short)0, c,
                                                false, false);
}

__device__ __forceinline__ float sigmoid_f(float x) {
  return 1.0f / (1.0f + __expf(-x));
}
__device__ __forceinline__ float tanh_f(float x) {
  x = fminf(fmaxf(x, -15.0f), 15.0f);
  float e = __expf(2.0f * x);
  return (e - 1.0f) / (e + 1.0f);
}

// Issue a TDM 2D tile load Global->LDS (D# per ISA 8.3/8.4).
//  data_size_log: 0=1B,1=2B,2=4B,3=8B. dim0/rows/stride0 in elements.
//  g1_extra: extra bits for group1 dword0 (pad_enable/interval/amount).
__device__ __forceinline__ void tdm_load_2d(unsigned lds_off, const void* gptr,
                                            unsigned data_size_log, unsigned dim0,
                                            unsigned rows, unsigned stride0,
                                            unsigned g1_extra) {
  const unsigned long long ga = (unsigned long long)(uintptr_t)gptr;
  v4u g0;
  g0[0] = 1u;                                   // count=1 (valid user D#)
  g0[1] = lds_off;                              // lds_addr (bytes)
  g0[2] = (unsigned)ga;                         // global_addr[31:0]
  g0[3] = (unsigned)((ga >> 32) & 0x1FFFFFFu)   // global_addr[56:32]
          | (2u << 30);                         // type = 2 ("image")
  v8i g1;
  g1[0] = (int)((data_size_log << 16) | g1_extra);
  g1[1] = (int)(dim0 << 16);                    // tensor_dim0[15:0]
  g1[2] = (int)(rows << 16);                    // tensor_dim0 hi | tensor_dim1 lo
  g1[3] = (int)(dim0 << 16);                    // tensor_dim1 hi | tile_dim0
  g1[4] = (int)rows;                            // tile_dim1 | tile_dim2=0
  g1[5] = (int)stride0;                         // tensor_dim0_stride[31:0]
  g1[6] = 0;                                    // stride0 hi | stride1 lo
  g1[7] = 0;                                    // stride1 hi
  v4i z4 = {};
  v8i z8 = {};
  // 6-arg form on this toolchain: (g0, g1, g2, g3, extra, cpol).
  __builtin_amdgcn_tensor_load_to_lds(g0, g1, z4, z4, z8, 0);
}

// ---------------------------------------------------------------- cvt ------
__global__ __launch_bounds__(256)
void cvt_f32_to_f16_kernel(const float* __restrict__ src, f16* __restrict__ dst,
                           int n) {
  int i = blockIdx.x * 256 + threadIdx.x;
  if (i < n) dst[i] = (f16)src[i];
}

// ------------------------------------------------------------- layer 0 -----
__global__ __launch_bounds__(256, 1)
void gru_layer0_kernel(const float* __restrict__ x,        // [B,S,7]
                       const float* __restrict__ wih_f, const float* __restrict__ bih_f,
                       const float* __restrict__ bhh_f,
                       const float* __restrict__ wih_b, const float* __restrict__ bih_b,
                       const float* __restrict__ bhh_b,
                       const f16*   __restrict__ whh16,    // [4][384][128]
                       f16*         __restrict__ h1_in)    // [B,S,256]
{
  const int dir   = blockIdx.x & 1;
  const int chunk = blockIdx.x >> 1;
  const int b0    = chunk * 16;
  const int tid   = threadIdx.x;
  const int wave  = tid >> 5, lane = tid & 31;
  const int col   = lane & 15, half = lane >> 4;
  const int colh  = wave * 16 + col;            // hidden column of this lane

  const float* wih = dir ? wih_b : wih_f;
  const float* bih = dir ? bih_b : bih_f;
  const float* bhh = dir ? bhh_b : bhh_f;
  const f16*   whh = whh16 + (size_t)dir * G3H * HH;   // layer-0 matrices 0,1

  __shared__ __attribute__((aligned(16))) f16   h16[16 * HS16];
  __shared__ __attribute__((aligned(16))) float h32[16 * HS32];
  __shared__ __attribute__((aligned(16))) float xst[2][16 * DIN];  // TDM tiles

  for (int i = tid; i < 16 * HS16; i += 256) h16[i] = (f16)0.0f;
  for (int i = tid; i < 16 * HS32; i += 256) h32[i] = 0.0f;

  // Per-lane input-gate weights (D_IN=7) and biases for gates r,z,n.
  float wr[3][DIN], bi[3], bh[3];
#pragma unroll
  for (int g = 0; g < 3; ++g) {
    const int n = g * HH + colh;
#pragma unroll
    for (int d = 0; d < DIN; ++d) wr[g][d] = wih[n * DIN + d];
    bi[g] = bih[n];
    bh[g] = bhh[n];
  }

  // W_hh B-fragments resident in VGPRs for the whole sequence.
  v16h wf[3][4];
#pragma unroll
  for (int g = 0; g < 3; ++g) {
    const int n = g * HH + colh;
#pragma unroll
    for (int kt = 0; kt < 4; ++kt)
      wf[g][kt] = *(const v16h*)(whh + (size_t)n * HH + kt * 32 + half * 16);
  }

  const float* gx_base = x + (size_t)b0 * SS * DIN;
  const unsigned xoff[2] = { (unsigned)(uintptr_t)&xst[0][0],
                             (unsigned)(uintptr_t)&xst[1][0] };
  const int t0 = dir ? (SS - 1) : 0;
  if (wave == 0)   // prime the pipeline: tile for t0 into buffer 0
    tdm_load_2d(xoff[0], gx_base + (size_t)t0 * DIN, 2, DIN, 16,
                (unsigned)(SS * DIN), 0);

  for (int si = 0; si < SS; ++si) {
    const int t = dir ? (SS - 1 - si) : si;

    if (wave == 0) __builtin_amdgcn_s_wait_tensorcnt(0);  // tile si landed
    __syncthreads();   // tile visible to all; prior h writes visible
    if (wave == 0 && si + 1 < SS) {                        // TDM tile si+1
      const int tn = dir ? (t - 1) : (t + 1);
      tdm_load_2d(xoff[(si + 1) & 1], gx_base + (size_t)tn * DIN, 2, DIN, 16,
                  (unsigned)(SS * DIN), 0);
    }
    const float* xt = xst[si & 1];

    // Recurrence gh = h . W_hh^T via WMMA first (head of the serial chain).
    v8f acc[3] = {};
#pragma unroll
    for (int kt = 0; kt < 4; ++kt) {
      union { v16h v; v8h h[2]; } a;
      const f16* ap = h16 + col * HS16 + kt * 32 + half * 16;
      a.h[0] = *(const v8h*)ap;
      a.h[1] = *(const v8h*)(ap + 8);
#pragma unroll
      for (int g = 0; g < 3; ++g)
        acc[g] = wmma_f32_16x16x32_f16(a.v, wf[g][kt], acc[g]);
    }

    // Input-gate contribution (rank-7 -> VALU, co-executes with WMMA).
    float gx[3][8];
#pragma unroll
    for (int v = 0; v < 8; ++v) {
      const int m = v + half * 8;
      float xv[DIN];
#pragma unroll
      for (int d = 0; d < DIN; ++d) xv[d] = xt[m * DIN + d];
#pragma unroll
      for (int g = 0; g < 3; ++g) {
        float a = bi[g];
#pragma unroll
        for (int d = 0; d < DIN; ++d) a = fmaf(wr[g][d], xv[d], a);
        gx[g][v] = a;
      }
    }
    __syncthreads();   // all reads of h (and xt) done before overwrite

    // Fused GRU gates; C element (m = v + half*8, n = col).
#pragma unroll
    for (int v = 0; v < 8; ++v) {
      const int m = v + half * 8;
      const float hprev = h32[m * HS32 + colh];
      const float r = sigmoid_f(gx[0][v] + acc[0][v] + bh[0]);
      const float z = sigmoid_f(gx[1][v] + acc[1][v] + bh[1]);
      const float nn = tanh_f(gx[2][v] + r * (acc[2][v] + bh[2]));
      const float hnew = (1.0f - z) * nn + z * hprev;
      h32[m * HS32 + colh] = hnew;
      const f16 hv = (f16)hnew;
      h16[m * HS16 + colh] = hv;
      h1_in[((size_t)(b0 + m) * SS + t) * C2H + dir * HH + colh] = hv;
    }
    // next iteration's top barrier orders h-write vs h-read
  }
}

// ------------------------------------------------------------- layer 1 -----
__global__ __launch_bounds__(256, 1)
void gru_layer1_kernel(const f16*   __restrict__ h1_in,   // [B,S,256]
                       const f16*   __restrict__ wih16,   // [2][384][256]
                       const float* __restrict__ bih_f, const float* __restrict__ bhh_f,
                       const float* __restrict__ bih_b, const float* __restrict__ bhh_b,
                       const f16*   __restrict__ whh16,   // [4][384][128]
                       f16*         __restrict__ gru_out) // [B,S,256]
{
  const int dir   = blockIdx.x & 1;
  const int chunk = blockIdx.x >> 1;
  const int b0    = chunk * 16;
  const int tid   = threadIdx.x;
  const int wave  = tid >> 5, lane = tid & 31;
  const int col   = lane & 15, half = lane >> 4;
  const int colh  = wave * 16 + col;

  const float* bih = dir ? bih_b : bih_f;
  const float* bhh = dir ? bhh_b : bhh_f;
  const f16*   whh = whh16 + (size_t)(2 + dir) * G3H * HH;  // layer-1 matrices
  const f16*   wih = wih16 + (size_t)dir * G3H * C2H;

  __shared__ __attribute__((aligned(16))) f16   h16[16 * HS16];
  __shared__ __attribute__((aligned(16))) float h32[16 * HS32];
  __shared__ __attribute__((aligned(16))) f16   xst[2][16 * XS1]; // TDM tiles

  for (int i = tid; i < 16 * HS16; i += 256) h16[i] = (f16)0.0f;
  for (int i = tid; i < 16 * HS32; i += 256) h32[i] = 0.0f;

  float bi[3], bh[3];
#pragma unroll
  for (int g = 0; g < 3; ++g) {
    bi[g] = bih[g * HH + colh];
    bh[g] = bhh[g * HH + colh];
  }

  v16h wf[3][4];   // W_hh fragments (K = 128)
  v16h wi[3][8];   // W_ih fragments (K = 256)
#pragma unroll
  for (int g = 0; g < 3; ++g) {
    const int n = g * HH + colh;
#pragma unroll
    for (int kt = 0; kt < 4; ++kt)
      wf[g][kt] = *(const v16h*)(whh + (size_t)n * HH + kt * 32 + half * 16);
#pragma unroll
    for (int kt = 0; kt < 8; ++kt)
      wi[g][kt] = *(const v16h*)(wih + (size_t)n * C2H + kt * 32 + half * 16);
  }

  // TDM with LDS padding: 4 DWORDs (16B) every 128 DWORDs (512B = one 256-f16
  // row) -> staged row stride XS1=264 f16, 4-bank skew, conflict-free b128s.
  const unsigned g1pad = (1u << 20) | (6u << 22) | (3u << 25);
  const f16* gx_base = h1_in + (size_t)b0 * SS * C2H;
  const unsigned xoff[2] = { (unsigned)(uintptr_t)&xst[0][0],
                             (unsigned)(uintptr_t)&xst[1][0] };
  const int t0 = dir ? (SS - 1) : 0;
  if (wave == 0)
    tdm_load_2d(xoff[0], gx_base + (size_t)t0 * C2H, 1, C2H, 16,
                (unsigned)(SS * C2H), g1pad);

  for (int si = 0; si < SS; ++si) {
    const int t = dir ? (SS - 1 - si) : si;

    if (wave == 0) __builtin_amdgcn_s_wait_tensorcnt(0);
    __syncthreads();
    if (wave == 0 && si + 1 < SS) {
      const int tn = dir ? (t - 1) : (t + 1);
      tdm_load_2d(xoff[(si + 1) & 1], gx_base + (size_t)tn * C2H, 1, C2H, 16,
                  (unsigned)(SS * C2H), g1pad);
    }
    const f16* xb = xst[si & 1];

    // Recurrence gh = h . W_hh^T first (head of the serial chain).
    v8f acc[3] = {};
#pragma unroll
    for (int kt = 0; kt < 4; ++kt) {
      union { v16h v; v8h h[2]; } a;
      const f16* ap = h16 + col * HS16 + kt * 32 + half * 16;
      a.h[0] = *(const v8h*)ap;
      a.h[1] = *(const v8h*)(ap + 8);
#pragma unroll
      for (int g = 0; g < 3; ++g)
        acc[g] = wmma_f32_16x16x32_f16(a.v, wf[g][kt], acc[g]);
    }

    // Input gates gx = x_t . W_ih^T via WMMA from the staged LDS tile.
    v8f gacc[3] = {};
#pragma unroll
    for (int kt = 0; kt < 8; ++kt) {
      union { v16h v; v8h h[2]; } a;
      const f16* ap = xb + col * XS1 + kt * 32 + half * 16;
      a.h[0] = *(const v8h*)ap;
      a.h[1] = *(const v8h*)(ap + 8);
#pragma unroll
      for (int g = 0; g < 3; ++g)
        gacc[g] = wmma_f32_16x16x32_f16(a.v, wi[g][kt], gacc[g]);
    }
    __syncthreads();   // h/xb reads complete

#pragma unroll
    for (int v = 0; v < 8; ++v) {
      const int m = v + half * 8;
      const float hprev = h32[m * HS32 + colh];
      const float r = sigmoid_f(gacc[0][v] + bi[0] + acc[0][v] + bh[0]);
      const float z = sigmoid_f(gacc[1][v] + bi[1] + acc[1][v] + bh[1]);
      const float nn = tanh_f(gacc[2][v] + bi[2] + r * (acc[2][v] + bh[2]));
      const float hnew = (1.0f - z) * nn + z * hprev;
      h32[m * HS32 + colh] = hnew;
      const f16 hv = (f16)hnew;
      h16[m * HS16 + colh] = hv;
      gru_out[((size_t)(b0 + m) * SS + t) * C2H + dir * HH + colh] = hv;
    }
  }
}

// ---------------------------------------------------------------- head -----
__global__ __launch_bounds__(256)
void head_kernel(const f16* __restrict__ gru_out,   // [B,S,256]
                 const float* __restrict__ wout,    // [256]
                 const float* __restrict__ bout,    // [1]
                 float* __restrict__ out)           // [B,S]
{
  const int wave = threadIdx.x >> 5, lane = threadIdx.x & 31;
  const size_t bs = (size_t)blockIdx.x * 8 + wave;
  const v8h hv = *(const v8h*)(gru_out + bs * C2H + lane * 8);
  const float* w = wout + lane * 8;
  float acc = 0.0f;
#pragma unroll
  for (int i = 0; i < 8; ++i) acc = fmaf((float)hv[i], w[i], acc);
#pragma unroll
  for (int off = 16; off >= 1; off >>= 1) acc += __shfl_xor(acc, off, 32);
  if (lane == 0) out[bs] = acc + bout[0];
}

// -------------------------------------------------------------- launch -----
extern "C" void kernel_launch(void* const* d_in, const int* in_sizes, int n_in,
                              void* d_out, int out_size, void* d_ws, size_t ws_size,
                              hipStream_t stream) {
  (void)in_sizes; (void)n_in; (void)out_size; (void)ws_size;

  const float* x     = (const float*)d_in[0];
  const float* wih0f = (const float*)d_in[1];
  const float* whh0f = (const float*)d_in[2];
  const float* bih0f = (const float*)d_in[3];
  const float* bhh0f = (const float*)d_in[4];
  const float* wih0b = (const float*)d_in[5];
  const float* whh0b = (const float*)d_in[6];
  const float* bih0b = (const float*)d_in[7];
  const float* bhh0b = (const float*)d_in[8];
  const float* wih1f = (const float*)d_in[9];
  const float* whh1f = (const float*)d_in[10];
  const float* bih1f = (const float*)d_in[11];
  const float* bhh1f = (const float*)d_in[12];
  const float* wih1b = (const float*)d_in[13];
  const float* whh1b = (const float*)d_in[14];
  const float* bih1b = (const float*)d_in[15];
  const float* bhh1b = (const float*)d_in[16];
  const float* wout  = (const float*)d_in[17];
  const float* bout  = (const float*)d_in[18];

  // Workspace layout (f16 elements): h1_in | gru_out | whh16[4] | wih1_16[2]
  f16* ws = (f16*)d_ws;
  const size_t n_act = (size_t)BB * SS * C2H;      // 67,108,864 per activation
  f16* h1_in   = ws;
  f16* gruout  = ws + n_act;
  f16* whh16   = ws + 2 * n_act;                   // 4 * 384*128
  f16* wih16   = whh16 + 4 * (size_t)G3H * HH;     // 2 * 384*256

  const int nhh = G3H * HH;   // 49152
  const int nih = G3H * C2H;  // 98304
  cvt_f32_to_f16_kernel<<<(nhh + 255) / 256, 256, 0, stream>>>(whh0f, whh16 + 0 * (size_t)nhh, nhh);
  cvt_f32_to_f16_kernel<<<(nhh + 255) / 256, 256, 0, stream>>>(whh0b, whh16 + 1 * (size_t)nhh, nhh);
  cvt_f32_to_f16_kernel<<<(nhh + 255) / 256, 256, 0, stream>>>(whh1f, whh16 + 2 * (size_t)nhh, nhh);
  cvt_f32_to_f16_kernel<<<(nhh + 255) / 256, 256, 0, stream>>>(whh1b, whh16 + 3 * (size_t)nhh, nhh);
  cvt_f32_to_f16_kernel<<<(nih + 255) / 256, 256, 0, stream>>>(wih1f, wih16 + 0 * (size_t)nih, nih);
  cvt_f32_to_f16_kernel<<<(nih + 255) / 256, 256, 0, stream>>>(wih1b, wih16 + 1 * (size_t)nih, nih);

  gru_layer0_kernel<<<8, 256, 0, stream>>>(x, wih0f, bih0f, bhh0f,
                                           wih0b, bih0b, bhh0b, whh16, h1_in);
  gru_layer1_kernel<<<8, 256, 0, stream>>>(h1_in, wih16, bih1f, bhh1f,
                                           bih1b, bhh1b, whh16, gruout);
  head_kernel<<<(BB * SS) / 8, 256, 0, stream>>>(gruout, wout, bout, (float*)d_out);
}
